// SwinTransformerBlock_30056181137847
// MI455X (gfx1250) — compile-verified, matrix operand order checked by reading, not run on hardware
//
#include <hip/hip_runtime.h>
#include <math.h>

// ---------------------------------------------------------------------------
// MI455X (gfx1250) implementation of the dual-stream 3D-Swin block.
// ~20 GFLOP / ~100MB touched => latency bound; everything L2-resident.
// All GEMMs use v_wmma_f32_16x16x32_f16 (wave32, 1 wave per 16x16 tile).
// ---------------------------------------------------------------------------

typedef __attribute__((ext_vector_type(16))) _Float16 v16h;
typedef __attribute__((ext_vector_type(8)))  _Float16 v8h;
typedef __attribute__((ext_vector_type(8)))  float    v8f;

#define HEADS 6
#define DIMC  96
#define CHC   48
#define HIDC  384
#define NTOK  343
#define NPAD  352          // 22 * 16
#define NWIN  125
#define GRD   35
#define LTOK  42875        // 35^3
#define LPAD  42880        // multiple of 16
#define MROWS (NWIN * NPAD)  // 44000
#define EPSLN 1e-5f

// ---------------- WMMA helpers (layouts per CDNA5 ISA 7.12.2) ---------------

__device__ __forceinline__ v8f wmma16(v16h a, v16h b, v8f c) {
  // D = A(16x32 f16) * B(32x16 f16) + C(16x16 f32)
  return __builtin_amdgcn_wmma_f32_16x16x32_f16(false, a, false, b, (short)0, c,
                                                false, false);
}

// A/B fragment from a row-major f16 matrix [rows][ld], K-slice of 32.
// lane (0..31): row = row0 + (lane&15); K indices: half h = lane>>4,
//   elems 0..7  -> k0 + h*8 + t        (VGPR 0..3)
//   elems 8..15 -> k0 + 16 + h*8 + t   (VGPR 4..7)
__device__ __forceinline__ v16h frag_ld32(const _Float16* __restrict__ p0,
                                          int ld, int row0, int k0) {
  int lane = threadIdx.x & 31;
  const _Float16* p =
      p0 + (size_t)(row0 + (lane & 15)) * ld + k0 + ((lane >> 4) << 3);
  v8h lo = *(const v8h*)p;
  v8h hi = *(const v8h*)(p + 16);
  v16h r;
#pragma unroll
  for (int t = 0; t < 8; ++t) { r[t] = lo[t]; r[8 + t] = hi[t]; }
  return r;
}

// Same but the matrix only has K=16 (head dim); upper 16 K's are zero.
__device__ __forceinline__ v16h frag_ld16(const _Float16* __restrict__ p0,
                                          int row0) {
  int lane = threadIdx.x & 31;
  const _Float16* p =
      p0 + (size_t)(row0 + (lane & 15)) * 16 + ((lane >> 4) << 3);
  v8h lo = *(const v8h*)p;
  v16h r = {};
#pragma unroll
  for (int t = 0; t < 8; ++t) r[t] = lo[t];
  return r;
}

// ---------------- weight convert: f32 [K][Nout] -> f16 [Nout][Kpad] ---------

__global__ void k_wt(const float* __restrict__ w, _Float16* __restrict__ wt,
                     int K, int Nout, int Kpad) {
  int i = blockIdx.x * blockDim.x + threadIdx.x;
  if (i >= Nout * Kpad) return;
  int n = i / Kpad, k = i % Kpad;
  wt[i] = (k < K) ? (_Float16)w[(size_t)k * Nout + n] : (_Float16)0.f;
}

// ------- LN1 + chessboard cross-stream shuffle + roll + window gather -------
// dest (w,n) <- source spatial q' = (win_pos + 3) mod 35 per dim.
__global__ void k_prep(const float* __restrict__ xa, const float* __restrict__ xb,
                       const float* __restrict__ g_a, const float* __restrict__ b_a,
                       const float* __restrict__ g_b, const float* __restrict__ b_b,
                       _Float16* __restrict__ xaw, _Float16* __restrict__ xbw) {
  int t = blockIdx.x * blockDim.x + threadIdx.x;
  if (t >= NWIN * NPAD) return;
  int w = t / NPAD, n = t % NPAD;
  _Float16* oa = xaw + (size_t)t * DIMC;
  _Float16* ob = xbw + (size_t)t * DIMC;
  if (n >= NTOK) {
    for (int c = 0; c < DIMC; ++c) { oa[c] = (_Float16)0.f; ob[c] = (_Float16)0.f; }
    return;
  }
  int wi = w / 25, wj = (w / 5) % 5, wk = w % 5;
  int i = n / 49, j = (n / 7) % 7, k = n % 7;
  int qh = (wi * 7 + i + 3) % GRD;
  int qw = (wj * 7 + j + 3) % GRD;
  int qt = (wk * 7 + k + 3) % GRD;
  int par = ((qh / 7) + (qw / 7) + (qt / 7)) & 1;   // chessboard parity of src win
  size_t flat = ((size_t)qh * GRD + qw) * GRD + qt;
  const float* pa = xa + flat * CHC;
  const float* pb = xb + flat * CHC;
  float la[CHC], lb[CHC];
  {
    float m = 0.f; for (int c = 0; c < CHC; ++c) m += pa[c]; m *= (1.f / CHC);
    float v = 0.f; for (int c = 0; c < CHC; ++c) { float d = pa[c] - m; v += d * d; }
    float r = rsqrtf(v * (1.f / CHC) + EPSLN);
    for (int c = 0; c < CHC; ++c) la[c] = (pa[c] - m) * r * g_a[c] + b_a[c];
  }
  {
    float m = 0.f; for (int c = 0; c < CHC; ++c) m += pb[c]; m *= (1.f / CHC);
    float v = 0.f; for (int c = 0; c < CHC; ++c) { float d = pb[c] - m; v += d * d; }
    float r = rsqrtf(v * (1.f / CHC) + EPSLN);
    for (int c = 0; c < CHC; ++c) lb[c] = (pb[c] - m) * r * g_b[c] + b_b[c];
  }
  for (int c = 0; c < CHC; ++c) {
    oa[c] = (_Float16)la[c];
    ob[c] = (_Float16)lb[c];
    oa[CHC + c] = (_Float16)(par == 0 ? lb[c] : la[c]);  // sbc stream
    ob[CHC + c] = (_Float16)(par == 0 ? la[c] : lb[c]);  // sac stream
  }
}

// ---------------- QKV GEMM: [MROWS,96] x [96,288] -> Q,K,Vt -----------------

__global__ __launch_bounds__(32)
void k_qkv(const _Float16* __restrict__ X, const _Float16* __restrict__ Wt,
           const float* __restrict__ bias,
           _Float16* __restrict__ Q,   // [w*6+h][NPAD][16], pre-scaled 0.25
           _Float16* __restrict__ K,   // [w*6+h][NPAD][16]
           _Float16* __restrict__ Vt)  // [w*6+h][16][NPAD]  (transposed)
{
  int mt = blockIdx.x, ct = blockIdx.y;
  int lane = threadIdx.x & 31, h = lane >> 4, ln = lane & 15;
  v8f acc = {};
#pragma unroll
  for (int kb = 0; kb < DIMC; kb += 32) {
    v16h a = frag_ld32(X, DIMC, mt * 16, kb);
    v16h b = frag_ld32(Wt, DIMC, ct * 16, kb);
    acc = wmma16(a, b, acc);
  }
  int col = ct * 16 + ln;
  float bv = bias[col];
  int sect = col / DIMC;       // uniform per tile (96 % 16 == 0)
  int cc = col % DIMC;
  int head = cc / 16, d = cc % 16;
#pragma unroll
  for (int jj = 0; jj < 8; ++jj) {
    int m = mt * 16 + jj + 8 * h;
    int w = m / NPAD, tok = m % NPAD;
    float val = acc[jj] + bv;
    size_t wh = (size_t)(w * HEADS + head);
    if (sect == 0)      Q[(wh * NPAD + tok) * 16 + d] = (_Float16)(val * 0.25f);
    else if (sect == 1) K[(wh * NPAD + tok) * 16 + d] = (_Float16)val;
    else                Vt[(wh * 16 + d) * NPAD + tok] = (_Float16)val;
  }
}

// ------- fused attention: QK^T (wmma) + rpb + mask -> LDS softmax -> PV -----

__global__ __launch_bounds__(32)
void k_attn(const _Float16* __restrict__ Q, const _Float16* __restrict__ K,
            const _Float16* __restrict__ Vt,
            const float* __restrict__ rpb,    // [2197][6]
            const float* __restrict__ mask,   // [125][343][343]
            _Float16* __restrict__ O)         // [MROWS][96]
{
  __shared__ float P[16 * NPAD];              // 22.5 KB score/prob panel
  int w = blockIdx.x / 22, rt = blockIdx.x % 22;
  int head = blockIdx.y;
  int lane = threadIdx.x & 31, h = lane >> 4, ln = lane & 15;
  size_t wh = (size_t)(w * HEADS + head);
  const _Float16* Qp = Q + wh * NPAD * 16;
  const _Float16* Kp = K + wh * NPAD * 16;
  const _Float16* Vp = Vt + wh * 16 * NPAD;
  int m0 = rt * 16;
  v16h qa = frag_ld16(Qp, m0);
  for (int nt = 0; nt < 22; ++nt) {
    v16h kb = frag_ld16(Kp, nt * 16);
    v8f s = {};
    s = wmma16(qa, kb, s);
#pragma unroll
    for (int jj = 0; jj < 8; ++jj) {
      int ml = m0 + jj + 8 * h;
      int nl = nt * 16 + ln;
      float sc;
      if (nl >= NTOK)      sc = -1e30f;      // pad columns: masked out
      else if (ml >= NTOK) sc = 0.f;         // pad rows: never read back
      else {
        int zm = ml / 49, ym = (ml / 7) % 7, xm = ml % 7;
        int zn = nl / 49, yn = (nl / 7) % 7, xn = nl % 7;
        int idx = (zm - zn + 6) * 169 + (ym - yn + 6) * 13 + (xm - xn + 6);
        sc = s[jj] + rpb[idx * HEADS + head] +
             mask[((size_t)w * NTOK + ml) * NTOK + nl];
      }
      P[(jj + 8 * h) * NPAD + nl] = sc;
    }
  }
  __syncthreads();
  if (lane < 16) {                            // row-wise softmax, 1 lane/row
    float* row = P + lane * NPAD;
    float mx = -1e30f;
    for (int n = 0; n < NTOK; ++n) mx = fmaxf(mx, row[n]);
    float sum = 0.f;
    for (int n = 0; n < NTOK; ++n) { float e = __expf(row[n] - mx); row[n] = e; sum += e; }
    float inv = 1.f / sum;
    for (int n = 0; n < NTOK; ++n) row[n] *= inv;
    for (int n = NTOK; n < NPAD; ++n) row[n] = 0.f;
  }
  __syncthreads();
  v8f acc = {};
  for (int kt = 0; kt < NPAD; kt += 32) {     // out = P @ V
    v16h pa;
    const float* row = P + ln * NPAD + kt + h * 8;
#pragma unroll
    for (int t = 0; t < 8; ++t) {
      pa[t] = (_Float16)row[t];
      pa[8 + t] = (_Float16)row[16 + t];
    }
    v16h vb = frag_ld32(Vp, NPAD, 0, kt);
    acc = wmma16(pa, vb, acc);
  }
#pragma unroll
  for (int jj = 0; jj < 8; ++jj) {
    int ml = m0 + jj + 8 * h;
    O[((size_t)w * NPAD + ml) * DIMC + head * 16 + ln] = (_Float16)acc[jj];
  }
}

// ------- proj (only the 48 surviving cols) + residual scatter (inverse map) -

__global__ __launch_bounds__(32)
void k_proj(const _Float16* __restrict__ O, const _Float16* __restrict__ Wt,
            const float* __restrict__ bias, const float* __restrict__ x0,
            float* __restrict__ x1) {
  int mt = blockIdx.x, ct = blockIdx.y;       // ct 0..2 (48 cols)
  int lane = threadIdx.x & 31, h = lane >> 4, ln = lane & 15;
  v8f acc = {};
#pragma unroll
  for (int kb = 0; kb < DIMC; kb += 32) {
    v16h a = frag_ld32(O, DIMC, mt * 16, kb);
    v16h b = frag_ld32(Wt, DIMC, ct * 16, kb);
    acc = wmma16(a, b, acc);
  }
  int col = ct * 16 + ln;
  float bv = bias[col];
#pragma unroll
  for (int jj = 0; jj < 8; ++jj) {
    int m = mt * 16 + jj + 8 * h;
    int w = m / NPAD, n = m % NPAD;
    if (n >= NTOK) continue;
    int wi = w / 25, wj = (w / 5) % 5, wk = w % 5;
    int i = n / 49, j = (n / 7) % 7, k = n % 7;
    int qh = (wi * 7 + i + 3) % GRD;
    int qw = (wj * 7 + j + 3) % GRD;
    int qt = (wk * 7 + k + 3) % GRD;
    size_t flat = ((size_t)qh * GRD + qw) * GRD + qt;
    x1[flat * CHC + col] = x0[flat * CHC + col] + acc[jj] + bv;
  }
}

// ---------------- LN2 -> f16 [LPAD][64] (K padded 48->64) -------------------

__global__ void k_ln2(const float* __restrict__ x1, const float* __restrict__ g,
                      const float* __restrict__ b, _Float16* __restrict__ y) {
  int t = blockIdx.x * blockDim.x + threadIdx.x;
  if (t >= LPAD) return;
  _Float16* o = y + (size_t)t * 64;
  if (t >= LTOK) { for (int c = 0; c < 64; ++c) o[c] = (_Float16)0.f; return; }
  const float* p = x1 + (size_t)t * CHC;
  float m = 0.f; for (int c = 0; c < CHC; ++c) m += p[c]; m *= (1.f / CHC);
  float v = 0.f; for (int c = 0; c < CHC; ++c) { float d = p[c] - m; v += d * d; }
  float r = rsqrtf(v * (1.f / CHC) + EPSLN);
  for (int c = 0; c < CHC; ++c) o[c] = (_Float16)((p[c] - m) * r * g[c] + b[c]);
  for (int c = CHC; c < 64; ++c) o[c] = (_Float16)0.f;
}

// ---------------- fc1: [LPAD,64] x [64,384] + GELU --------------------------

__global__ __launch_bounds__(32)
void k_fc1(const _Float16* __restrict__ X, const _Float16* __restrict__ Wt,
           const float* __restrict__ bias, _Float16* __restrict__ H) {
  int mt = blockIdx.x, ct = blockIdx.y;
  int lane = threadIdx.x & 31, h = lane >> 4, ln = lane & 15;
  v8f acc = {};
#pragma unroll
  for (int kb = 0; kb < 64; kb += 32) {
    v16h a = frag_ld32(X, 64, mt * 16, kb);
    v16h b = frag_ld32(Wt, 64, ct * 16, kb);
    acc = wmma16(a, b, acc);
  }
  int col = ct * 16 + ln;
  float bv = bias[col];
#pragma unroll
  for (int jj = 0; jj < 8; ++jj) {
    int m = mt * 16 + jj + 8 * h;
    float v = acc[jj] + bv;
    float gel = 0.5f * v * (1.f + erff(v * 0.70710678118654752f));  // exact GELU
    H[(size_t)m * HIDC + col] = (_Float16)gel;
  }
}

// ---------------- fc2: [LPAD,384] x [384,48] + residual ---------------------

__global__ __launch_bounds__(32)
void k_fc2(const _Float16* __restrict__ H, const _Float16* __restrict__ Wt,
           const float* __restrict__ bias, const float* __restrict__ x1,
           float* __restrict__ x2) {
  int mt = blockIdx.x, ct = blockIdx.y;       // ct 0..2
  int lane = threadIdx.x & 31, h = lane >> 4, ln = lane & 15;
  v8f acc = {};
#pragma unroll
  for (int kb = 0; kb < HIDC; kb += 32) {
    v16h a = frag_ld32(H, HIDC, mt * 16, kb);
    v16h b = frag_ld32(Wt, HIDC, ct * 16, kb);
    acc = wmma16(a, b, acc);
  }
  int col = ct * 16 + ln;
  float bv = bias[col];
#pragma unroll
  for (int jj = 0; jj < 8; ++jj) {
    int m = mt * 16 + jj + 8 * h;
    if (m < LTOK)
      x2[(size_t)m * CHC + col] = acc[jj] + bv + x1[(size_t)m * CHC + col];
  }
}

// ---------------- gate: g = sigmoid(relu(mean(xa2) @ gw1) @ gw2 + gb2) ------

__global__ void k_gate(const float* __restrict__ xa2, const float* __restrict__ gw1,
                       const float* __restrict__ gw2, const float* __restrict__ gb2,
                       float* __restrict__ gout) {
  __shared__ float colsum[CHC];
  float loc[CHC];
  for (int c = 0; c < CHC; ++c) loc[c] = 0.f;
  for (int r = threadIdx.x; r < LTOK; r += blockDim.x) {
    const float* p = xa2 + (size_t)r * CHC;
    for (int c = 0; c < CHC; ++c) loc[c] += p[c];
  }
  if (threadIdx.x < CHC) colsum[threadIdx.x] = 0.f;
  __syncthreads();
  for (int c = 0; c < CHC; ++c) atomicAdd(&colsum[c], loc[c]);
  __syncthreads();
  if (threadIdx.x == 0) {
    float z = gb2[0];
    for (int c2 = 0; c2 < CHC / 4; ++c2) {
      float a = 0.f;
      for (int c = 0; c < CHC; ++c)
        a += (colsum[c] * (1.f / LTOK)) * gw1[c * (CHC / 4) + c2];
      z += fmaxf(a, 0.f) * gw2[c2];
    }
    gout[0] = 1.f / (1.f + __expf(-z));
  }
}

// ---------------- final mix: out_a = xa2 + g*xb2 ; out_b = xb2 --------------

__global__ void k_final(const float* __restrict__ xa2, const float* __restrict__ xb2,
                        const float* __restrict__ g, float* __restrict__ out) {
  size_t i = (size_t)blockIdx.x * blockDim.x + threadIdx.x;
  if (i >= (size_t)LTOK * CHC) return;
  float gv = g[0];
  out[i] = xa2[i] + gv * xb2[i];
  out[(size_t)LTOK * CHC + i] = xb2[i];
}

// ---------------------------------------------------------------------------

extern "C" void kernel_launch(void* const* d_in, const int* in_sizes, int n_in,
                              void* d_out, int out_size, void* d_ws, size_t ws_size,
                              hipStream_t stream) {
  (void)in_sizes; (void)n_in; (void)out_size; (void)ws_size;
  const float* x_a    = (const float*)d_in[0];
  const float* x_b    = (const float*)d_in[1];
  const float* maskm  = (const float*)d_in[2];
  const float* n1ag   = (const float*)d_in[3];
  const float* n1ab   = (const float*)d_in[4];
  const float* n1bg   = (const float*)d_in[5];
  const float* n1bb   = (const float*)d_in[6];
  const float* rpb_a  = (const float*)d_in[7];
  const float* qkvw_a = (const float*)d_in[8];
  const float* qkvb_a = (const float*)d_in[9];
  const float* projw_a= (const float*)d_in[10];
  const float* projb_a= (const float*)d_in[11];
  const float* rpb_b  = (const float*)d_in[12];
  const float* qkvw_b = (const float*)d_in[13];
  const float* qkvb_b = (const float*)d_in[14];
  const float* projw_b= (const float*)d_in[15];
  const float* projb_b= (const float*)d_in[16];
  const float* n2ag   = (const float*)d_in[17];
  const float* n2ab   = (const float*)d_in[18];
  const float* n2bg   = (const float*)d_in[19];
  const float* n2bb   = (const float*)d_in[20];
  const float* fc1w_a = (const float*)d_in[21];
  const float* fc1b_a = (const float*)d_in[22];
  const float* fc2w_a = (const float*)d_in[23];
  const float* fc2b_a = (const float*)d_in[24];
  const float* fc1w_b = (const float*)d_in[25];
  const float* fc1b_b = (const float*)d_in[26];
  const float* fc2w_b = (const float*)d_in[27];
  const float* fc2b_b = (const float*)d_in[28];
  const float* gw1    = (const float*)d_in[29];
  const float* gw2    = (const float*)d_in[30];
  const float* gb2    = (const float*)d_in[31];

  // ---- workspace layout (with lifetime-based aliasing, ~118 MB total) ----
  char* W = (char*)d_ws;
  const size_t SZ = (size_t)MROWS * DIMC * 2;   // 8,448,000 (== per-stream QKV buf)
  _Float16* xaw = (_Float16*)(W + 0 * SZ);
  _Float16* xbw = (_Float16*)(W + 1 * SZ);
  _Float16* Qa  = (_Float16*)(W + 2 * SZ);
  _Float16* Ka  = (_Float16*)(W + 3 * SZ);
  _Float16* Vta = (_Float16*)(W + 4 * SZ);
  _Float16* Qb  = (_Float16*)(W + 5 * SZ);
  _Float16* Kb  = (_Float16*)(W + 6 * SZ);
  _Float16* Vtb = (_Float16*)(W + 7 * SZ);
  _Float16* Oa  = (_Float16*)(W + 8 * SZ);
  _Float16* Ob  = (_Float16*)(W + 9 * SZ);
  const size_t B0 = 10 * SZ;                    // 84,480,000
  const size_t SX = 8232192;                    // LTOK*48*4 rounded to 256
  float* x1a = (float*)(W + B0 + 0 * SX);
  float* x1b = (float*)(W + B0 + 1 * SX);
  float* x2a = (float*)(W + B0 + 2 * SX);
  float* x2b = (float*)(W + B0 + 3 * SX);
  float* gsc = (float*)(W + B0 + 4 * SX);
  char*  WT  = W + B0 + 4 * SX + 256;
  // aliases (regions dead by the time these are written):
  _Float16* Ha   = (_Float16*)(W + 0 * SZ);     // over xaw..Ka   (32.9 MB)
  _Float16* Hb   = (_Float16*)(W + 4 * SZ);     // over Vta..Vtb  (32.9 MB)
  _Float16* mina = (_Float16*)(W + 8 * SZ);     // over Oa        ( 5.5 MB)
  _Float16* minb = (_Float16*)(W + 9 * SZ);     // over Ob
  // f16 weights (transposed to [Nout][Kpad]):
  _Float16* qkvwt_a = (_Float16*)(WT);            // 288*96
  _Float16* qkvwt_b = qkvwt_a + 288 * 96;
  _Float16* projwt_a= qkvwt_b + 288 * 96;         // 96*96
  _Float16* projwt_b= projwt_a + 96 * 96;
  _Float16* fc1wt_a = projwt_b + 96 * 96;         // 384*64
  _Float16* fc1wt_b = fc1wt_a + 384 * 64;
  _Float16* fc2wt_a = fc1wt_b + 384 * 64;         // 48*384
  _Float16* fc2wt_b = fc2wt_a + 48 * 384;

  // ---- 0) weight packing ----
  k_wt<<<(288*96 + 255)/256, 256, 0, stream>>>(qkvw_a, qkvwt_a, 96, 288, 96);
  k_wt<<<(288*96 + 255)/256, 256, 0, stream>>>(qkvw_b, qkvwt_b, 96, 288, 96);
  k_wt<<<(96*96  + 255)/256, 256, 0, stream>>>(projw_a, projwt_a, 96, 96, 96);
  k_wt<<<(96*96  + 255)/256, 256, 0, stream>>>(projw_b, projwt_b, 96, 96, 96);
  k_wt<<<(384*64 + 255)/256, 256, 0, stream>>>(fc1w_a, fc1wt_a, 48, 384, 64);
  k_wt<<<(384*64 + 255)/256, 256, 0, stream>>>(fc1w_b, fc1wt_b, 48, 384, 64);
  k_wt<<<(48*384 + 255)/256, 256, 0, stream>>>(fc2w_a, fc2wt_a, 384, 48, 384);
  k_wt<<<(48*384 + 255)/256, 256, 0, stream>>>(fc2w_b, fc2wt_b, 384, 48, 384);

  // ---- 1) LN1 + chessboard + roll + window gather -> f16 ----
  k_prep<<<(NWIN*NPAD + 127)/128, 128, 0, stream>>>(x_a, x_b, n1ag, n1ab,
                                                    n1bg, n1bb, xaw, xbw);

  // ---- 2) QKV GEMMs (WMMA) ----
  dim3 gq(MROWS / 16, 18);
  k_qkv<<<gq, 32, 0, stream>>>(xaw, qkvwt_a, qkvb_a, Qa, Ka, Vta);
  k_qkv<<<gq, 32, 0, stream>>>(xbw, qkvwt_b, qkvb_b, Qb, Kb, Vtb);

  // ---- 3) fused attention (WMMA + LDS softmax) ----
  dim3 ga(NWIN * 22, HEADS);
  k_attn<<<ga, 32, 0, stream>>>(Qa, Ka, Vta, rpb_a, maskm, Oa);
  k_attn<<<ga, 32, 0, stream>>>(Qb, Kb, Vtb, rpb_b, maskm, Ob);

  // ---- 4) proj (48 cols) + residual scatter ----
  dim3 gp(MROWS / 16, 3);
  k_proj<<<gp, 32, 0, stream>>>(Oa, projwt_a, projb_a, x_a, x1a);
  k_proj<<<gp, 32, 0, stream>>>(Ob, projwt_b, projb_b, x_b, x1b);

  // ---- 5) LN2 ----
  k_ln2<<<(LPAD + 127)/128, 128, 0, stream>>>(x1a, n2ag, n2ab, mina);
  k_ln2<<<(LPAD + 127)/128, 128, 0, stream>>>(x1b, n2bg, n2bb, minb);

  // ---- 6) fc1 + GELU (WMMA) ----
  dim3 g1(LPAD / 16, 24);
  k_fc1<<<g1, 32, 0, stream>>>(mina, fc1wt_a, fc1b_a, Ha);
  k_fc1<<<g1, 32, 0, stream>>>(minb, fc1wt_b, fc1b_b, Hb);

  // ---- 7) fc2 + residual (WMMA) ----
  dim3 g2(LPAD / 16, 3);
  k_fc2<<<g2, 32, 0, stream>>>(Ha, fc2wt_a, fc2b_a, x1a, x2a);
  k_fc2<<<g2, 32, 0, stream>>>(Hb, fc2wt_b, fc2b_b, x1b, x2b);

  // ---- 8) gate + final mix ----
  k_gate<<<1, 256, 0, stream>>>(x2a, gw1, gw2, gb2, gsc);
  k_final<<<((size_t)LTOK*CHC + 255)/256, 256, 0, stream>>>(x2a, x2b, gsc,
                                                            (float*)d_out);
}
